// Ntm_55052890800717
// MI455X (gfx1250) — compile-verified, matrix operand order checked by reading, not run on hardware
//
#include <hip/hip_runtime.h>
#include <hip/hip_bf16.h>

typedef __attribute__((ext_vector_type(16))) _Float16 v16h;
typedef __attribute__((ext_vector_type(8)))  _Float16 v8h;
typedef __attribute__((ext_vector_type(8)))  float    v8f;

// global (addrspace(1)) fragment pointers -> force global_load_b128 (LOADcnt only),
// decoupled from the LDS A-fragment ds_load stream (DScnt).
typedef const __attribute__((address_space(1))) _Float16 GF;
typedef const __attribute__((address_space(1))) v16h     GV16;

#define CTRL 256
#define WORD 128
#define MEMN 128
#define DIN  64
#define BB   512
#define TT   512
#define GIN  192            /* DIN + WORD */
#define NG   768            /* 3*CTRL */
#define EPSV 1e-6f

#define ROWS 16             /* batch rows per workgroup */
#define NWG  (BB / ROWS)    /* 32 persistent workgroups */
#define NTHREADS 512        /* 16 waves (wave32) */

#define IH_NT 48
#define IH_KS 6
#define HH_NT 48
#define HH_KS 8
#define HD_NT 8
#define HD_KS 8

#define SMEM_BYTES 161856

// Defeat LICM: treat the uniform pointer as redefined each iteration so the
// loop-invariant fragment loads cannot be hoisted out of the t-loop (and spilled).
#define OPAQUE_PTR(p) asm volatile("" : "+s"(p))

// ---------------------------------------------------------------- helpers
__device__ __forceinline__ float sigmoidf_(float x) {
    return 1.0f / (1.0f + __expf(-x));
}
__device__ __forceinline__ float wred_sum(float v) {
    #pragma unroll
    for (int o = 16; o > 0; o >>= 1) v += __shfl_xor(v, o, 32);
    return v;
}
__device__ __forceinline__ float wred_max(float v) {
    #pragma unroll
    for (int o = 16; o > 0; o >>= 1) v = fmaxf(v, __shfl_xor(v, o, 32));
    return v;
}

// A fragment (16xK tile, K-slab ks) from a row-major f16 activation in LDS.
// Lane L: m = L%16; halves 0..7 -> k = (L/16)*8 + 0..7, halves 8..15 -> +16.
__device__ __forceinline__ v16h load_a(const _Float16* act, int stride, int ks, int lane) {
    const int m  = lane & 15;
    const int hi = (lane >> 4) << 3;
    const _Float16* p = act + m * stride + ks * 32 + hi;
    v8h lo = *(const v8h*)(p);
    v8h hh = *(const v8h*)(p + 16);
    v16h a;
    #pragma unroll
    for (int i = 0; i < 8; ++i) { a[i] = lo[i]; a[i + 8] = hh[i]; }
    return a;
}

// One 16x16 output tile: acc = bias; for ks: acc += A(act) * B(frag)
__device__ __forceinline__ v8f mm_tile(const _Float16* act, int stride, int kslabs,
                                       GF* frag, int nt, float bv, int lane) {
    v8f acc;
    #pragma unroll
    for (int i = 0; i < 8; ++i) acc[i] = bv;
    for (int ks = 0; ks < kslabs; ++ks) {
        v16h a = load_a(act, stride, ks, lane);
        v16h b = *(GV16*)(frag + ((size_t)(nt * kslabs + ks) * 32 + lane) * 16);
        acc = __builtin_amdgcn_wmma_f32_16x16x32_f16(false, a, false, b,
                                                     (short)0, acc, false, false);
    }
    return acc;
}

__device__ __forceinline__ void store_tile(float* out, int ostride, int nt, int lane, v8f acc) {
    const int n  = lane & 15;
    const int mb = (lane >> 4) << 3;
    #pragma unroll
    for (int v = 0; v < 8; ++v) out[(mb + v) * ostride + nt * 16 + n] = acc[v];
}

// ---------------------------------------------------------------- prep kernels
// Swizzle W [Nout,Kin] row-major f32 -> WMMA B fragments, f16:
// dst[(nt*kslabs+ks)*512 + lane*16 + h] = W[nt*16 + lane%16, ks*32 + kofs(h,lane)]
__global__ void k_prep_frags(const float* __restrict__ W, _Float16* __restrict__ dst,
                             int ntiles, int kslabs, int kin) {
    int idx = blockIdx.x * blockDim.x + threadIdx.x;
    int total = ntiles * kslabs * 512;
    if (idx >= total) return;
    int frag = idx >> 9;
    int rem  = idx & 511;
    int ln   = rem >> 4;
    int h    = rem & 15;
    int nt = frag / kslabs;
    int ks = frag - nt * kslabs;
    int n  = nt * 16 + (ln & 15);
    int k  = ks * 32 + (h & 7) + ((h >> 3) << 4) + ((ln >> 4) << 3);
    dst[idx] = (_Float16)W[n * kin + k];
}

// MT[r][w][j] = M0[j][w]  (transposed, broadcast over batch)
__global__ void k_prep_M(const float* __restrict__ M0, float* __restrict__ MT) {
    int idx = blockIdx.x * blockDim.x + threadIdx.x;
    if (idx >= BB * WORD * MEMN) return;
    int j = idx & 127;
    int w = (idx >> 7) & 127;
    MT[idx] = M0[j * WORD + w];
}

// len[r] = #steps with batch_sizes[t] > r (packed-batch active length)
__global__ void k_prep_len(const int* __restrict__ bs, int* __restrict__ lenArr) {
    int r = blockIdx.x * blockDim.x + threadIdx.x;
    if (r >= BB) return;
    int c = 0;
    for (int t = 0; t < TT; ++t) c += (bs[t] > r) ? 1 : 0;
    lenArr[r] = c;
}

// ---------------------------------------------------------------- main persistent kernel
__global__ __launch_bounds__(NTHREADS, 1)
void ntm_persist(const float* __restrict__ data,
                 const float* __restrict__ b_ih, const float* __restrict__ b_hh,
                 const float* __restrict__ b_key, const float* __restrict__ Wbeta,
                 const float* __restrict__ b_beta,
                 const float* __restrict__ b_erase, const float* __restrict__ b_add,
                 const _Float16* __restrict__ fIH, const _Float16* __restrict__ fHH,
                 const _Float16* __restrict__ fKEY, const _Float16* __restrict__ fERASE,
                 const _Float16* __restrict__ fADD,
                 const int* __restrict__ lenArr,
                 float* __restrict__ MT,
                 float* __restrict__ gH, float* __restrict__ gR) {
    extern __shared__ char smem[];
    float*    sH    = (float*)(smem);                 // 16 x 256 f32   16384
    _Float16* sHh   = (_Float16*)(smem + 16384);      // 16 x 256 f16    8192
    _Float16* sINh  = (_Float16*)(smem + 24576);      // 16 x 192 f16    6144
    float*    sRead = (float*)(smem + 30720);         // 16 x 128 f32    8192
    float*    sGI   = (float*)(smem + 38912);         // 16 x 768 f32   49152
    float*    sGH   = (float*)(smem + 88064);         // 16 x 768 f32   49152
    float*    sK    = (float*)(smem + 137216);        // 16 x 128 f32    8192
    float*    sE    = (float*)(smem + 145408);        //                 8192
    float*    sA    = (float*)(smem + 153600);        //                 8192
    int*      sLen  = (int*)(smem + 161792);          //                   64

    const int tid  = threadIdx.x;
    const int lane = tid & 31;
    const int wave = tid >> 5;
    const int row0 = blockIdx.x * ROWS;

    for (int i = tid; i < ROWS * CTRL; i += NTHREADS) { sH[i] = 0.f; sHh[i] = (_Float16)0.f; }
    for (int i = tid; i < ROWS * WORD; i += NTHREADS) sRead[i] = 0.f;
    if (tid < ROWS) sLen[tid] = lenArr[row0 + tid];
    __syncthreads();

    const int    myLen   = sLen[wave];
    const size_t myMbase = (size_t)(row0 + wave) * (WORD * MEMN);

    for (int t = 0; t < TT; ++t) {
        // Per-iteration opaque, addrspace(1)-pinned fragment pointers:
        // forces fresh global_load_b128 each step (L2-resident shared weights),
        // no hoist/spill, no DScnt coupling.
        GF* fIHt    = (GF*)(unsigned long long)fIH;
        GF* fHHt    = (GF*)(unsigned long long)fHH;
        GF* fKEYt   = (GF*)(unsigned long long)fKEY;
        GF* fERASEt = (GF*)(unsigned long long)fERASE;
        GF* fADDt   = (GF*)(unsigned long long)fADD;
        OPAQUE_PTR(fIHt);
        OPAQUE_PTR(fHHt);
        OPAQUE_PTR(fKEYt);
        OPAQUE_PTR(fERASEt);
        OPAQUE_PTR(fADDt);

        // ---- (a) build ctrl_in = [x_t, read] in f16
        for (int i = tid; i < ROWS * GIN; i += NTHREADS) {
            int m = i / GIN, c = i - m * GIN;
            float v = (c < DIN)
                ? data[(size_t)t * BB * DIN + (size_t)(row0 + m) * DIN + c]
                : sRead[m * WORD + (c - DIN)];
            sINh[i] = (_Float16)v;
            if (c < DIN && t + 1 < TT)
                __builtin_prefetch(&data[(size_t)(t + 1) * BB * DIN + (size_t)(row0 + m) * DIN + c], 0, 1);
        }
        __syncthreads();

        // ---- (b) gi = ctrl_in @ W_ih^T + b_ih   (48 tiles over 16 waves)
        #pragma unroll
        for (int tt = 0; tt < 3; ++tt) {
            int nt = wave + tt * 16;
            v8f acc = mm_tile(sINh, GIN, IH_KS, fIHt, nt, b_ih[nt * 16 + (lane & 15)], lane);
            store_tile(sGI, NG, nt, lane, acc);
        }
        // ---- (c) gh = h @ W_hh^T + b_hh
        #pragma unroll
        for (int tt = 0; tt < 3; ++tt) {
            int nt = wave + tt * 16;
            v8f acc = mm_tile(sHh, CTRL, HH_KS, fHHt, nt, b_hh[nt * 16 + (lane & 15)], lane);
            store_tile(sGH, NG, nt, lane, acc);
        }
        __syncthreads();

        // ---- (d) GRU gate fusion + masked h update
        for (int i = tid; i < ROWS * CTRL; i += NTHREADS) {
            int m = i >> 8, c = i & 255;
            float ir = sGI[m * NG + c], iz = sGI[m * NG + c + CTRL], in_ = sGI[m * NG + c + 2 * CTRL];
            float hr = sGH[m * NG + c], hz = sGH[m * NG + c + CTRL], hn  = sGH[m * NG + c + 2 * CTRL];
            float r = sigmoidf_(ir + hr);
            float z = sigmoidf_(iz + hz);
            float n = tanhf(in_ + r * hn);
            float hp = sH[i];
            float hnew = (t < sLen[m]) ? ((1.f - z) * n + z * hp) : hp;
            sH[i]  = hnew;
            sHh[i] = (_Float16)hnew;
        }
        __syncthreads();

        // ---- (e) head projections: key/erase/add = 24 tiles over 16 waves
        for (int tt = wave; tt < 24; tt += 16) {
            int which = tt >> 3, nt = tt & 7;
            GF* frag = (which == 0) ? fKEYt : (which == 1) ? fERASEt : fADDt;
            const float* bias = (which == 0) ? b_key : (which == 1) ? b_erase : b_add;
            v8f acc = mm_tile(sHh, CTRL, HD_KS, frag, nt, bias[nt * 16 + (lane & 15)], lane);
            float* dst = (which == 0) ? sK : (which == 1) ? sE : sA;
            const int n = lane & 15, mb = (lane >> 4) << 3;
            #pragma unroll
            for (int v = 0; v < 8; ++v) {
                float x = acc[v];
                if (which == 0) x = tanhf(x);
                else if (which == 1) x = sigmoidf_(x);
                dst[(mb + v) * WORD + nt * 16 + n] = x;
            }
        }
        __syncthreads();

        // ---- (f) memory addressing + read + erase/add write (wave owns row m=wave)
        if (t < myLen) {
            const int m = wave;
            // beta = softplus(h . Wbeta + b)
            float bp = 0.f;
            #pragma unroll
            for (int q = 0; q < 8; ++q)
                bp += sH[m * CTRL + lane * 8 + q] * Wbeta[lane * 8 + q];
            float bsum = wred_sum(bp) + b_beta[0];
            float beta = (bsum > 20.f) ? bsum : log1pf(__expf(bsum));
            // ||k||
            float kq = 0.f;
            #pragma unroll
            for (int q = 0; q < 4; ++q) { float kv = sK[m * WORD + lane * 4 + q]; kq += kv * kv; }
            kq = wred_sum(kq);
            const float scale = beta / (sqrtf(kq) + EPSV);

            // pass 1: dot(M_j,k) and ||M_j|| streaming MT[r][w][j] (lanes own j)
            float d0 = 0, d1 = 0, d2 = 0, d3 = 0, n0 = 0, n1 = 0, n2 = 0, n3 = 0;
            const float4* Mp = (const float4*)(MT + myMbase);
            for (int w = 0; w < WORD; ++w) {
                float  kv = sK[m * WORD + w];
                float4 mv = Mp[w * 32 + lane];
                d0 += mv.x * kv; d1 += mv.y * kv; d2 += mv.z * kv; d3 += mv.w * kv;
                n0 += mv.x * mv.x; n1 += mv.y * mv.y; n2 += mv.z * mv.z; n3 += mv.w * mv.w;
            }
            float l0 = d0 * scale / (sqrtf(n0) + EPSV);
            float l1 = d1 * scale / (sqrtf(n1) + EPSV);
            float l2 = d2 * scale / (sqrtf(n2) + EPSV);
            float l3 = d3 * scale / (sqrtf(n3) + EPSV);
            float mx = wred_max(fmaxf(fmaxf(l0, l1), fmaxf(l2, l3)));
            float e0 = __expf(l0 - mx), e1 = __expf(l1 - mx);
            float e2 = __expf(l2 - mx), e3 = __expf(l3 - mx);
            float inv = 1.f / wred_sum(e0 + e1 + e2 + e3);
            float w0 = e0 * inv, w1 = e1 * inv, w2 = e2 * inv, w3 = e3 * inv;

            // pass 2: read (old M) + erase/add write
            float4* Mw = (float4*)(MT + myMbase);
            for (int w = 0; w < WORD; ++w) {
                float4 mv = Mw[w * 32 + lane];
                float rp = wred_sum(mv.x * w0 + mv.y * w1 + mv.z * w2 + mv.w * w3);
                if (lane == 0) sRead[m * WORD + w] = rp;
                float ev = sE[m * WORD + w], av = sA[m * WORD + w];
                mv.x = mv.x * (1.f - w0 * ev) + w0 * av;
                mv.y = mv.y * (1.f - w1 * ev) + w1 * av;
                mv.z = mv.z * (1.f - w2 * ev) + w2 * av;
                mv.w = mv.w * (1.f - w3 * ev) + w3 * av;
                Mw[w * 32 + lane] = mv;
            }
        }
        __syncthreads();
    }

    for (int i = tid; i < ROWS * CTRL; i += NTHREADS) gH[(size_t)row0 * CTRL + i] = sH[i];
    for (int i = tid; i < ROWS * WORD; i += NTHREADS) gR[(size_t)row0 * WORD + i] = sRead[i];
}

// ---------------------------------------------------------------- final gather (unsort)
__global__ void k_gather(const float* __restrict__ gH, const float* __restrict__ gR,
                         const int* __restrict__ unsort, float* __restrict__ out) {
    int i = blockIdx.x * blockDim.x + threadIdx.x;
    if (i < BB * CTRL) {
        int b = i / CTRL, c = i - b * CTRL;
        out[i] = gH[unsort[b] * CTRL + c];
    } else if (i < BB * CTRL + BB * WORD) {
        int ii = i - BB * CTRL;
        int b = ii / WORD, c = ii - b * WORD;
        out[i] = gR[unsort[b] * WORD + c];
    }
}

// ---------------------------------------------------------------- launch
extern "C" void kernel_launch(void* const* d_in, const int* in_sizes, int n_in,
                              void* d_out, int out_size, void* d_ws, size_t ws_size,
                              hipStream_t stream) {
    const float* data    = (const float*)d_in[0];
    const int*   bs      = (const int*)d_in[1];
    const int*   unsort  = (const int*)d_in[2];
    const float* W_ih    = (const float*)d_in[3];
    const float* b_ih    = (const float*)d_in[4];
    const float* W_hh    = (const float*)d_in[5];
    const float* b_hh    = (const float*)d_in[6];
    const float* W_key   = (const float*)d_in[7];
    const float* b_key   = (const float*)d_in[8];
    const float* W_beta  = (const float*)d_in[9];
    const float* b_beta  = (const float*)d_in[10];
    const float* W_erase = (const float*)d_in[11];
    const float* b_erase = (const float*)d_in[12];
    const float* W_add   = (const float*)d_in[13];
    const float* b_add   = (const float*)d_in[14];
    const float* M0      = (const float*)d_in[15];

    char* ws = (char*)d_ws;
    size_t off = 0;
    auto carve = [&](size_t bytes) -> char* {
        char* p = ws + off;
        off += (bytes + 255) & ~(size_t)255;
        return p;
    };
    float*    MT     = (float*)carve((size_t)BB * WORD * MEMN * 4);   // 33.5 MB
    _Float16* fIH    = (_Float16*)carve((size_t)IH_NT * IH_KS * 512 * 2);
    _Float16* fHH    = (_Float16*)carve((size_t)HH_NT * HH_KS * 512 * 2);
    _Float16* fKEY   = (_Float16*)carve((size_t)HD_NT * HD_KS * 512 * 2);
    _Float16* fERASE = (_Float16*)carve((size_t)HD_NT * HD_KS * 512 * 2);
    _Float16* fADD   = (_Float16*)carve((size_t)HD_NT * HD_KS * 512 * 2);
    int*      lenArr = (int*)carve(BB * 4);
    float*    gH     = (float*)carve((size_t)BB * CTRL * 4);
    float*    gR     = (float*)carve((size_t)BB * WORD * 4);
    (void)in_sizes; (void)n_in; (void)out_size; (void)ws_size;

    // weight swizzle + state init (cheap; rerun every call for determinism)
    {
        int n1 = IH_NT * IH_KS * 512;
        k_prep_frags<<<(n1 + 255) / 256, 256, 0, stream>>>(W_ih, fIH, IH_NT, IH_KS, GIN);
        int n2 = HH_NT * HH_KS * 512;
        k_prep_frags<<<(n2 + 255) / 256, 256, 0, stream>>>(W_hh, fHH, HH_NT, HH_KS, CTRL);
        int n3 = HD_NT * HD_KS * 512;
        k_prep_frags<<<(n3 + 255) / 256, 256, 0, stream>>>(W_key,   fKEY,   HD_NT, HD_KS, CTRL);
        k_prep_frags<<<(n3 + 255) / 256, 256, 0, stream>>>(W_erase, fERASE, HD_NT, HD_KS, CTRL);
        k_prep_frags<<<(n3 + 255) / 256, 256, 0, stream>>>(W_add,   fADD,   HD_NT, HD_KS, CTRL);
        int nm = BB * WORD * MEMN;
        k_prep_M<<<(nm + 255) / 256, 256, 0, stream>>>(M0, MT);
        k_prep_len<<<2, 256, 0, stream>>>(bs, lenArr);
    }

    ntm_persist<<<NWG, NTHREADS, SMEM_BYTES, stream>>>(
        data, b_ih, b_hh, b_key, W_beta, b_beta, b_erase, b_add,
        fIH, fHH, fKEY, fERASE, fADD, lenArr, MT, gH, gR);

    int ng = BB * CTRL + BB * WORD;
    k_gather<<<(ng + 255) / 256, 256, 0, stream>>>(gH, gR, unsort, (float*)d_out);
}
